// RelativeMultiHeadAttentionBlock_39118562132417
// MI455X (gfx1250) — matrix operand order repaired and checked by reading, not compiled, not run on hardware
//
#include <hip/hip_runtime.h>

typedef unsigned short u16t;
typedef unsigned int   u32t;
typedef __attribute__((ext_vector_type(16))) __bf16 v16bf;
typedef __attribute__((ext_vector_type(8)))  float  v8f;
typedef __attribute__((ext_vector_type(4)))  u32t   v4u;
typedef __attribute__((ext_vector_type(8)))  int    vi8t;
typedef __attribute__((ext_vector_type(4)))  int    vi4t;

namespace {
constexpr int Bx = 2, Tx = 1024, Mx = 1024, Fx = 1024, Nx = 16, Hx = 64;
constexpr int Sx = Mx + Tx;   // 2048 keys
constexpr int NH = Nx * Hx;   // 1024
}

// ---------------- helpers ----------------

__device__ __forceinline__ u16t f2bf(float x) {
  u32t b = __float_as_uint(x);
  b += 0x7FFFu + ((b >> 16) & 1u);   // round-to-nearest-even
  return (u16t)(b >> 16);
}

__device__ __forceinline__ v16bf load_frag16(const u16t* p0, const u16t* p1) {
  union { v16bf v; v4u q[2]; } f;
  f.q[0] = *(const v4u*)p0;
  f.q[1] = *(const v4u*)p1;
  return f.v;
}

__device__ __forceinline__ v16bf zero_frag16() {
  union { v16bf v; v4u q[2]; } f;
  v4u z = {0u, 0u, 0u, 0u};
  f.q[0] = z; f.q[1] = z;
  return f.v;
}

__device__ __forceinline__ v8f zero8() {
  v8f z = {0.f, 0.f, 0.f, 0.f, 0.f, 0.f, 0.f, 0.f};
  return z;
}

__device__ __forceinline__ v8f wmma_bf16(v16bf a, v16bf b, v8f c) {
  // D = A(16x32 bf16) x B(32x16 bf16) + C(16x16 f32)
  return __builtin_amdgcn_wmma_f32_16x16x32_bf16(false, a, false, b, (short)0, c, false, false);
}

__device__ __forceinline__ float rmax16(float v) {
#pragma unroll
  for (int off = 1; off < 16; off <<= 1) v = fmaxf(v, __shfl_xor(v, off, 32));
  return v;
}
__device__ __forceinline__ float rsum16(float v) {
#pragma unroll
  for (int off = 1; off < 16; off <<= 1) v += __shfl_xor(v, off, 32);
  return v;
}

// ---------------- conversion kernels ----------------

__global__ void k_cvt_bf16(const float* __restrict__ src, u16t* __restrict__ dst, int n) {
  int i = blockIdx.x * blockDim.x + threadIdx.x;
  if (i < n) dst[i] = f2bf(src[i]);
}

// WT[c * rows + r] = bf16(W[r * cols + c])   (W is rows x cols row-major)
__global__ void k_transpose_bf16(const float* __restrict__ W, u16t* __restrict__ WT,
                                 int rows, int cols) {
  int i = blockIdx.x * blockDim.x + threadIdx.x;
  if (i >= rows * cols) return;
  int r = i / cols, c = i % cols;
  WT[(size_t)c * rows + r] = f2bf(W[i]);
}

__global__ void k_concat_kv(const float* __restrict__ mem, const float* __restrict__ x,
                            u16t* __restrict__ kvb) {
  int i = blockIdx.x * blockDim.x + threadIdx.x;
  if (i >= Bx * Sx * Fx) return;
  int f = i % Fx;
  int s = (i / Fx) % Sx;
  int b = i / (Fx * Sx);
  float v = (s < Mx) ? mem[((size_t)b * Mx + s) * Fx + f]
                     : x[((size_t)b * Tx + (s - Mx)) * Fx + f];
  kvb[i] = f2bf(v);
}

// ki[b][s] = episode_idx[b][s] for s<M; qi[b][t]=episode_idx[b][M-1]+cumsum(dones) for s>=M
__global__ void k_build_ids(const int* __restrict__ ep, const int* __restrict__ dn,
                            int* __restrict__ ki) {
  int b = threadIdx.x;
  if (b >= Bx) return;
  for (int s = 0; s < Mx; ++s) ki[b * Sx + s] = ep[b * Mx + s];
  int acc = ep[b * Mx + Mx - 1];
  for (int t = 0; t < Tx; ++t) { acc += dn[b * Tx + t]; ki[b * Sx + Mx + t] = acc; }
}

// ---------------- generic bf16 WMMA GEMM (K = 1024, Ncols = 1024) ----------------
// C[Mr x 1024] = A[Mr x 1024] * WT^T + bias ;  WT stored [col][k] row-major.
// One wave computes a 16x64 tile.  Ping-pong double buffering (2 k-steps per
// iteration): loads for step k+32 issue before the WMMAs of step k, and the
// buffers alternate so no register copies are needed.
__global__ void __launch_bounds__(256) k_gemm_bf16(
    const u16t* __restrict__ A, const u16t* __restrict__ WT,
    const float* __restrict__ bias, float* __restrict__ outF,
    u16t* __restrict__ outH, int Mr) {
  const int lane = threadIdx.x & 31;
  const int wave = blockIdx.x * (blockDim.x >> 5) + (threadIdx.x >> 5);
  const int r0 = (wave >> 4) * 16;      // 16 column tiles of 64 per row band
  const int n0 = (wave & 15) * 64;
  if (r0 >= Mr) return;
  const int ln = lane & 15, lh = lane >> 4;
  const int ah = lh << 3, bh = lh << 3;

  const u16t* Ap = A + (size_t)(r0 + ln) * 1024;
  const u16t* Bp[4];
#pragma unroll
  for (int c = 0; c < 4; ++c) Bp[c] = WT + (size_t)(n0 + c * 16 + ln) * 1024;

  v8f acc[4];
#pragma unroll
  for (int c = 0; c < 4; ++c) acc[c] = zero8();

  // prologue: buffer 0 holds fragments for k = 0
  v16bf a0 = load_frag16(Ap + ah, Ap + 16 + ah);
  v16bf b0[4];
#pragma unroll
  for (int c = 0; c < 4; ++c) b0[c] = load_frag16(Bp[c] + bh, Bp[c] + bh + 16);
  v16bf a1, b1[4];

  for (int k = 0; k < 1024 - 64; k += 64) {
    // prefetch k+32 into buffer 1, then consume buffer 0
    a1 = load_frag16(Ap + (k + 32) + ah, Ap + (k + 32) + 16 + ah);
#pragma unroll
    for (int c = 0; c < 4; ++c)
      b1[c] = load_frag16(Bp[c] + (k + 32) + bh, Bp[c] + (k + 32) + bh + 16);
#pragma unroll
    for (int c = 0; c < 4; ++c) acc[c] = wmma_bf16(a0, b0[c], acc[c]);
    // prefetch k+64 into buffer 0, then consume buffer 1
    a0 = load_frag16(Ap + (k + 64) + ah, Ap + (k + 64) + 16 + ah);
#pragma unroll
    for (int c = 0; c < 4; ++c)
      b0[c] = load_frag16(Bp[c] + (k + 64) + bh, Bp[c] + (k + 64) + bh + 16);
#pragma unroll
    for (int c = 0; c < 4; ++c) acc[c] = wmma_bf16(a1, b1[c], acc[c]);
  }
  // epilogue: k = 960 (buffer 0) and k = 992 (buffer 1)
  a1 = load_frag16(Ap + 992 + ah, Ap + 992 + 16 + ah);
#pragma unroll
  for (int c = 0; c < 4; ++c) b1[c] = load_frag16(Bp[c] + 992 + bh, Bp[c] + 992 + bh + 16);
#pragma unroll
  for (int c = 0; c < 4; ++c) acc[c] = wmma_bf16(a0, b0[c], acc[c]);
#pragma unroll
  for (int c = 0; c < 4; ++c) acc[c] = wmma_bf16(a1, b1[c], acc[c]);

  const int rbs = lh << 3;
#pragma unroll
  for (int c = 0; c < 4; ++c) {
    int col = n0 + c * 16 + ln;
    float bv = bias ? bias[col] : 0.f;
#pragma unroll
    for (int r = 0; r < 8; ++r) {
      size_t idx = (size_t)(r0 + rbs + r) * 1024 + col;
      float v = acc[c][r] + bv;
      if (outF) outF[idx] = v; else outH[idx] = f2bf(v);
    }
  }
}

// ---------------- fused flash attention with relative bias ----------------
// Grid: B*N*(T/64) blocks x 128 threads (4 waves; wave w owns queries t0..t0+15).
// bias[t,s] = q_v[t] . r[s - t + T - 1]  (valid exactly where causal mask holds).
// V tile (the transposed operand) is staged global->LDS by the Tensor Data
// Mover (tensor_load_to_lds, TENSORcnt), then transposed LDS->LDS.
__global__ void __launch_bounds__(128) k_flash(
    const u16t* __restrict__ qub, const u16t* __restrict__ qvb,
    const u16t* __restrict__ kb, const u16t* __restrict__ vb,
    const u16t* __restrict__ rb, const int* __restrict__ ki,
    u16t* __restrict__ attnb) {
  __shared__ alignas(16) u16t vstage[64 * 64];   // V tile, natural [s][h] (TDM dest)
  __shared__ alignas(16) u16t vtT[64 * 64];      // V^T: [h][s_local]
  __shared__ float Gt[4][16 * 80];               // per-wave band bias
  __shared__ alignas(16) u16t pt[4][16 * 64];    // per-wave probs (bf16)

  const int tid = threadIdx.x;
  const int lane = tid & 31;
  const int w = tid >> 5;
  const int ln = lane & 15, lh = lane >> 4;
  const int ah = lh << 3, bh = lh << 3, rbs = lh << 3;
  const float scale = 0.125f;                    // 1/sqrt(64)

  const int tpb = Tx / 64;                       // 16
  const int b    = blockIdx.x / (Nx * tpb);
  const int rem  = blockIdx.x % (Nx * tpb);
  const int n    = rem / tpb;
  const int tblk = (rem % tpb) * 64;
  const int t0   = tblk + w * 16;

  // Q fragments (A layout) straight from global
  const u16t* qup = qub + ((size_t)(b * Tx + t0 + ln)) * NH + n * Hx;
  const u16t* qvp = qvb + ((size_t)(b * Tx + t0 + ln)) * NH + n * Hx;
  v16bf qu_f[2], qv_f[2];
#pragma unroll
  for (int ks = 0; ks < 2; ++ks) {
    qu_f[ks] = load_frag16(qup + ks * 32 + ah, qup + ks * 32 + 16 + ah);
    qv_f[ks] = load_frag16(qvp + ks * 32 + ah, qvp + ks * 32 + 16 + ah);
  }

  int q_ids[8];
#pragma unroll
  for (int r = 0; r < 8; ++r) q_ids[r] = ki[b * Sx + Mx + t0 + rbs + r];

  float m_run[8], l_run[8];
  v8f O[4];
#pragma unroll
  for (int h = 0; h < 4; ++h) O[h] = zero8();
#pragma unroll
  for (int r = 0; r < 8; ++r) { m_run[r] = -1e30f; l_run[r] = 0.f; }

  const int s_endv = Mx + tblk + 64;             // causal bound for this block (<= Sx)
  for (int s0 = 0; s0 < s_endv; s0 += 64) {
    __syncthreads();
#if __has_builtin(__builtin_amdgcn_tensor_load_to_lds) && __has_builtin(__builtin_amdgcn_s_wait_tensorcnt)
    // --- TDM: stage 64x64 bf16 V tile global->LDS (no VGPR traffic) ---
    if (w == 0) {
      unsigned lds_off = (unsigned)(size_t)(void*)vstage;
      unsigned long long ga =
          (unsigned long long)(size_t)(const void*)(vb + ((size_t)(b * Sx + s0)) * NH + n * Hx);
      // D# group0: [1:0]=count=1; [63:32]=lds_addr; [120:64]=global_addr; [127:126]=type=2
      v4u g0 = { 1u, lds_off, (u32t)ga,
                 ((u32t)(ga >> 32) & 0x01FFFFFFu) | 0x80000000u };
      // D# group1: data_size=1(2B); dim0=64; dim1=64; tile0=64; tile1=64; stride0=1024
      vi8t g1 = { (int)0x10000, (int)(64u << 16), (int)(64u << 16), (int)(64u << 16),
                  64, 1024, 0, 0 };
      vi4t gz4 = { 0, 0, 0, 0 };
      vi8t gz8 = { 0, 0, 0, 0, 0, 0, 0, 0 };
      __builtin_amdgcn_tensor_load_to_lds(g0, g1, gz4, gz4, gz8, 0);
      __builtin_amdgcn_s_wait_tensorcnt(0);
    }
    __syncthreads();
    // transpose LDS->LDS: vstage[s][h] -> vtT[h][s]
    for (int i = tid; i < 512; i += 128) {
      int sl = i >> 3, h8 = (i & 7) << 3;
      union { v4u q; u16t u[8]; } f;
      f.q = *(const v4u*)(vstage + sl * 64 + h8);
#pragma unroll
      for (int j = 0; j < 8; ++j) vtT[(h8 + j) * 64 + sl] = f.u[j];
    }
#else
    // fallback: global -> VGPR -> transposed LDS
    for (int i = tid; i < 512; i += 128) {
      int sl = i >> 3, h8 = (i & 7) << 3;
      union { v4u q; u16t u[8]; } f;
      f.q = *(const v4u*)(vb + ((size_t)(b * Sx + s0 + sl)) * NH + n * Hx + h8);
#pragma unroll
      for (int j = 0; j < 8; ++j) vtT[(h8 + j) * 64 + sl] = f.u[j];
    }
#endif
    __syncthreads();

    // band bias G[m][j] = q_v[t0+m] . r[dlow + j], j in [0,80)
    const int dlow = s0 - t0 + Tx - 16;
#pragma unroll
    for (int g = 0; g < 5; ++g) {
      v8f acc = zero8();
      int drow = dlow + g * 16 + ln;
#pragma unroll
      for (int ks = 0; ks < 2; ++ks) {
        v16bf bf;
        if (drow < Sx) {
          const u16t* rp = rb + (size_t)drow * NH + n * Hx + ks * 32 + bh;
          bf = load_frag16(rp, rp + 16);
        } else {
          bf = zero_frag16();
        }
        acc = wmma_bf16(qv_f[ks], bf, acc);
      }
#pragma unroll
      for (int r = 0; r < 8; ++r) Gt[w][(rbs + r) * 80 + g * 16 + ln] = acc[r];
    }
    __syncthreads();

    // QK + bias + mask + online softmax, per 16-key chunk
#pragma unroll
    for (int c = 0; c < 4; ++c) {
      v8f acc = zero8();
      int srow = s0 + c * 16 + ln;
      const u16t* kp = kb + ((size_t)(b * Sx + srow)) * NH + n * Hx;
#pragma unroll
      for (int ks = 0; ks < 2; ++ks)
        acc = wmma_bf16(qu_f[ks], load_frag16(kp + ks * 32 + bh, kp + ks * 32 + bh + 16), acc);
      int kiv = ki[b * Sx + srow];
      float vals[8];
#pragma unroll
      for (int r = 0; r < 8; ++r) {
        int m = rbs + r;
        float g = Gt[w][m * 80 + (c * 16 + ln - m + 15)];
        float vv = scale * acc[r] + scale * g;
        bool cond = (srow <= Mx + t0 + m) && (kiv == q_ids[r]);
        vals[r] = cond ? vv : -1e30f;
      }
#pragma unroll
      for (int r = 0; r < 8; ++r) {
        float mx = rmax16(vals[r]);
        float mnew = fmaxf(m_run[r], mx);
        float alpha = __expf(m_run[r] - mnew);
        float p = (vals[r] > -1e29f) ? __expf(vals[r] - mnew) : 0.f;
        float ps = rsum16(p);
        l_run[r] = l_run[r] * alpha + ps;
        m_run[r] = mnew;
#pragma unroll
        for (int h = 0; h < 4; ++h) O[h][r] *= alpha;
        pt[w][(rbs + r) * 64 + c * 16 + ln] = f2bf(p);
      }
    }
    __syncthreads();

    // O += P x V
    const u16t* pp = &pt[w][ln * 64];
#pragma unroll
    for (int ks = 0; ks < 2; ++ks) {
      v16bf pf = load_frag16(pp + ks * 32 + ah, pp + ks * 32 + 16 + ah);
#pragma unroll
      for (int h = 0; h < 4; ++h) {
        const u16t* vp = &vtT[(h * 16 + ln) * 64 + ks * 32 + bh];
        O[h] = wmma_bf16(pf, load_frag16(vp, vp + 16), O[h]);
      }
    }
  }

  // normalize and store attn (bf16) at [b, t, n*64 + h]
#pragma unroll
  for (int r = 0; r < 8; ++r) {
    float inv = 1.f / l_run[r];
    size_t row = (size_t)(b * Tx + t0 + rbs + r) * NH + n * Hx;
#pragma unroll
    for (int h = 0; h < 4; ++h) attnb[row + h * 16 + ln] = f2bf(O[h][r] * inv);
  }
}

// ---------------- host launch ----------------

extern "C" void kernel_launch(void* const* d_in, const int* in_sizes, int n_in,
                              void* d_out, int out_size, void* d_ws, size_t ws_size,
                              hipStream_t stream) {
  (void)in_sizes; (void)n_in; (void)out_size; (void)ws_size;
  const float* x    = (const float*)d_in[0];
  const float* rel  = (const float*)d_in[1];
  const float* mem  = (const float*)d_in[2];
  const int*   ep   = (const int*)d_in[3];
  const int*   dn   = (const int*)d_in[4];
  const float* Wq   = (const float*)d_in[5];
  const float* Wk   = (const float*)d_in[6];
  const float* Wv   = (const float*)d_in[7];
  const float* Wr   = (const float*)d_in[8];
  const float* u    = (const float*)d_in[9];
  const float* v    = (const float*)d_in[10];
  const float* Wout = (const float*)d_in[11];
  const float* bout = (const float*)d_in[12];
  float* out = (float*)d_out;

  char* wsp = (char*)d_ws;
  size_t off = 0;
  auto alloc = [&](size_t bytes) -> void* {
    void* p = wsp + off;
    off += (bytes + 255) & ~(size_t)255;
    return p;
  };
  u16t* xb    = (u16t*)alloc((size_t)Bx * Tx * Fx * 2);
  u16t* kvb   = (u16t*)alloc((size_t)Bx * Sx * Fx * 2);
  u16t* relb  = (u16t*)alloc((size_t)Sx * Fx * 2);
  u16t* WqT   = (u16t*)alloc((size_t)Fx * NH * 2);
  u16t* WkT   = (u16t*)alloc((size_t)Fx * NH * 2);
  u16t* WvT   = (u16t*)alloc((size_t)Fx * NH * 2);
  u16t* WrT   = (u16t*)alloc((size_t)Fx * NH * 2);
  u16t* WoT   = (u16t*)alloc((size_t)NH * Fx * 2);
  u16t* qub   = (u16t*)alloc((size_t)Bx * Tx * NH * 2);
  u16t* qvb   = (u16t*)alloc((size_t)Bx * Tx * NH * 2);
  u16t* kbuf  = (u16t*)alloc((size_t)Bx * Sx * NH * 2);
  u16t* vbuf  = (u16t*)alloc((size_t)Bx * Sx * NH * 2);
  u16t* rbuf  = (u16t*)alloc((size_t)Sx * NH * 2);
  u16t* attnb = (u16t*)alloc((size_t)Bx * Tx * NH * 2);
  int*  kid   = (int*)alloc((size_t)Bx * Sx * 4);

  const int TB = 256;
  k_cvt_bf16<<<(Bx * Tx * Fx + TB - 1) / TB, TB, 0, stream>>>(x, xb, Bx * Tx * Fx);
  k_concat_kv<<<(Bx * Sx * Fx + TB - 1) / TB, TB, 0, stream>>>(mem, x, kvb);
  k_cvt_bf16<<<(Sx * Fx + TB - 1) / TB, TB, 0, stream>>>(rel, relb, Sx * Fx);
  k_transpose_bf16<<<(Fx * NH + TB - 1) / TB, TB, 0, stream>>>(Wq, WqT, Fx, NH);
  k_transpose_bf16<<<(Fx * NH + TB - 1) / TB, TB, 0, stream>>>(Wk, WkT, Fx, NH);
  k_transpose_bf16<<<(Fx * NH + TB - 1) / TB, TB, 0, stream>>>(Wv, WvT, Fx, NH);
  k_transpose_bf16<<<(Fx * NH + TB - 1) / TB, TB, 0, stream>>>(Wr, WrT, Fx, NH);
  k_transpose_bf16<<<(NH * Fx + TB - 1) / TB, TB, 0, stream>>>(Wout, WoT, NH, Fx);
  k_build_ids<<<1, 32, 0, stream>>>(ep, dn, kid);

  auto gemm = [&](const u16t* A, const u16t* WT, const float* bias,
                  float* oF, u16t* oH, int Mr) {
    int tiles = (Mr / 16) * 16;                  // 16 col-tiles of 64
    k_gemm_bf16<<<tiles / 8, 256, 0, stream>>>(A, WT, bias, oF, oH, Mr);
  };
  gemm(xb,   WqT, u,       nullptr, qub,  Bx * Tx);   // q_u = xWq + u
  gemm(xb,   WqT, v,       nullptr, qvb,  Bx * Tx);   // q_v = xWq + v
  gemm(kvb,  WkT, nullptr, nullptr, kbuf, Bx * Sx);   // k
  gemm(kvb,  WvT, nullptr, nullptr, vbuf, Bx * Sx);   // v
  gemm(relb, WrT, nullptr, nullptr, rbuf, Sx);        // r

  k_flash<<<Bx * Nx * (Tx / 64), 128, 0, stream>>>(qub, qvb, kbuf, vbuf, rbuf, kid, attnb);

  gemm(attnb, WoT, bout, out, nullptr, Bx * Tx);      // out = attn @ Wout + b_out
}
